// SelfSEAttention_30331059044903
// MI455X (gfx1250) — matrix operand order repaired and checked by reading, not compiled, not run on hardware
//
#include <hip/hip_runtime.h>

typedef __attribute__((ext_vector_type(16))) __bf16 v16bf;
typedef __attribute__((ext_vector_type(8)))  float  v8f;
typedef __attribute__((ext_vector_type(4)))  int    v4i;

#define C_DIM 256
#define CQ    32
#define NPIX  4096
#define BATCH 4

#define AS1 __attribute__((address_space(1)))
#define AS3 __attribute__((address_space(3)))

#if defined(__has_builtin)
#if __has_builtin(__builtin_amdgcn_global_load_async_to_lds_b128)
#define USE_ASYNC_LDS 1
#endif
#endif

__device__ __forceinline__ unsigned short f2bf(float f) {
  unsigned u = __float_as_uint(f);
  u += 0x7FFFu + ((u >> 16) & 1u);      // round to nearest even
  return (unsigned short)(u >> 16);
}

// stage 16 bytes global -> LDS
__device__ __forceinline__ void stage_b128(const unsigned short* g, unsigned short* l) {
#ifdef USE_ASYNC_LDS
  __builtin_amdgcn_global_load_async_to_lds_b128((AS1 v4i*)g, (AS3 v4i*)l, 0, 0);
#else
  *(uint4*)l = *(const uint4*)g;
#endif
}

__device__ __forceinline__ void stage_wait() {
#ifdef USE_ASYNC_LDS
  asm volatile("s_wait_asynccnt 0x0" ::: "memory");
#endif
}

union FragB { v16bf v; unsigned u[8]; };

// ---------------- projections ----------------
__global__ __launch_bounds__(256)
void proj_qk_kernel(const float* __restrict__ x,
                    const float* __restrict__ Wq, const float* __restrict__ bq,
                    const float* __restrict__ Wk, const float* __restrict__ bk,
                    unsigned short* __restrict__ qw, unsigned short* __restrict__ kw)
{
  int t = blockIdx.x * 256 + threadIdx.x;      // B*N*CQ threads
  int c = t & (CQ - 1);
  int n = (t >> 5) & (NPIX - 1);
  int b = t >> 17;
  const float* xp = x + (size_t)b * C_DIM * NPIX + n;
  const float* wq = Wq + c * C_DIM;
  const float* wk = Wk + c * C_DIM;
  float sq = 0.f, sk = 0.f;
  for (int j = 0; j < C_DIM; ++j) {
    float xv = xp[(size_t)j * NPIX];
    sq += wq[j] * xv;
    sk += wk[j] * xv;
  }
  qw[t] = f2bf(sq + bq[c]);   // layout [b][n][c]
  kw[t] = f2bf(sk + bk[c]);   // layout [b][n][c]
}

__global__ __launch_bounds__(256)
void proj_v_kernel(const float* __restrict__ x,
                   const float* __restrict__ Wv, const float* __restrict__ bv,
                   unsigned short* __restrict__ vw)
{
  int t = blockIdx.x * 256 + threadIdx.x;      // B*C*N threads
  int n = t & (NPIX - 1);
  int o = (t >> 12) & (C_DIM - 1);
  int b = t >> 20;
  const float* xp = x + (size_t)b * C_DIM * NPIX + n;
  const float* w  = Wv + o * C_DIM;
  float s = 0.f;
  for (int j = 0; j < C_DIM; ++j) s += w[j] * xp[(size_t)j * NPIX];
  vw[t] = f2bf(s + bv[o]);    // layout [b][o][n]
}

// ---------------- fused flash attention + residual + mean accumulation ----------------
// block = 8 waves; wave owns 16 query rows; N-tile = 64 per iteration
__global__ __launch_bounds__(256)
void attn_kernel(const unsigned short* __restrict__ qw,
                 const unsigned short* __restrict__ kw,
                 const unsigned short* __restrict__ vw,
                 const float* __restrict__ x,
                 const float* __restrict__ gamma,
                 float* __restrict__ self_out,
                 float* __restrict__ mean_acc)
{
  __shared__ __align__(16) unsigned short Qs[128 * CQ];   // 8 KB
  __shared__ __align__(16) unsigned short Ks[64 * CQ];    // 4 KB  [n][c]
  __shared__ __align__(16) unsigned short Vs[C_DIM * 64]; // 32 KB [c][n]
  __shared__ __align__(16) float          Ss[8][16 * 64]; // 32 KB per-wave scores
  __shared__ __align__(16) unsigned short Ps[8][16 * 64]; // 16 KB per-wave probs (bf16)
  __shared__ float rowScale[8][16];
  __shared__ float rowLs[8][16];

  const int tid  = threadIdx.x;
  const int wave = tid >> 5;
  const int lane = tid & 31;
  const int half = lane >> 4;
  const int l16  = lane & 15;
  const int b      = blockIdx.y;
  const int mBlock = blockIdx.x * 128;

  // stage Q tile [128 x 32] bf16
  {
    int row = tid >> 1;
    int off = (tid & 1) * 16;                      // ushort offset
    const uint4* g = (const uint4*)(qw + ((size_t)(b * NPIX + mBlock + row)) * CQ + off);
    uint4* s = (uint4*)(Qs + row * CQ + off);
    s[0] = g[0]; s[1] = g[1];
  }
  __syncthreads();

  // A fragment of Q (16x32 bf16): lane half selects K groups {0-7,16-23} / {8-15,24-31}
  FragB aQ;
  #pragma unroll
  for (int v = 0; v < 8; ++v) {
    int kb = ((v < 4) ? 2 * v : 2 * v + 8) + 8 * half;
    aQ.u[v] = *(const unsigned*)(Qs + (wave * 16 + l16) * CQ + kb);
  }

  v8f acc[16];
  #pragma unroll
  for (int t = 0; t < 16; ++t) { v8f z = {}; acc[t] = z; }

  float m_i = -__builtin_inff();   // valid in lanes 0-15
  float l_i = 0.f;

  for (int n0 = 0; n0 < NPIX; n0 += 64) {
    __syncthreads();
    // stage K tile [64 x 32] (n-major): 1 x 16B per thread (async -> LDS)
    {
      int row = tid >> 2;
      int off = (tid & 3) * 8;                     // ushort offset
      stage_b128(kw + ((size_t)(b * NPIX + n0 + row)) * CQ + off, Ks + row * CQ + off);
    }
    // stage V tile [256 x 64] (c-major): 8 x 16B per thread (one 128B row, async -> LDS)
    {
      const unsigned short* src = vw + ((size_t)(b * C_DIM + tid)) * NPIX + n0;
      unsigned short* dst = Vs + tid * 64;
      #pragma unroll
      for (int j = 0; j < 8; ++j) stage_b128(src + j * 8, dst + j * 8);
      if (n0 + 64 < NPIX)
        __builtin_prefetch(src + 64, 0, 3);
    }
    stage_wait();
    __syncthreads();

    // S = Q (16x32) x K^T (32x64): four bf16 WMMAs, K-dim == Cq == 32
    v8f sfr[4];
    #pragma unroll
    for (int st = 0; st < 4; ++st) {
      FragB bK;
      #pragma unroll
      for (int v = 0; v < 8; ++v) {
        int kb = ((v < 4) ? 2 * v : 2 * v + 8) + 8 * half; // B-layout: K groups {0-7,16-23} / {8-15,24-31} by half
        bK.u[v] = *(const unsigned*)(Ks + (st * 16 + l16) * CQ + kb);
      }
      v8f zc = {};
      sfr[st] = __builtin_amdgcn_wmma_f32_16x16x32_bf16(false, aQ.v, false, bK.v, (short)0, zc, false, false);
    }

    // scatter S (C/D layout: M = r + 8*half, N = l16) into per-wave LDS
    #pragma unroll
    for (int st = 0; st < 4; ++st) {
      #pragma unroll
      for (int r = 0; r < 8; ++r) {
        int m = r + 8 * half;
        Ss[wave][m * 64 + st * 16 + l16] = sfr[st][r];
      }
    }
    __syncthreads();

    // online softmax: lanes 0-15 each own one row of 64 scores
    if (lane < 16) {
      const float* srow = &Ss[wave][lane * 64];
      float tmax = srow[0];
      #pragma unroll
      for (int j = 1; j < 64; ++j) tmax = fmaxf(tmax, srow[j]);
      float newm  = fmaxf(m_i, tmax);
      float scale = __expf(m_i - newm);
      float sum = 0.f;
      unsigned short* prow = &Ps[wave][lane * 64];
      #pragma unroll
      for (int j = 0; j < 64; ++j) {
        float p = __expf(srow[j] - newm);
        sum += p;
        prow[j] = f2bf(p);
      }
      l_i = l_i * scale + sum;
      m_i = newm;
      rowScale[wave][lane] = scale;
    }
    __syncthreads();

    // rescale running output accumulators (once per 64 columns)
    float fs[8];
    #pragma unroll
    for (int r = 0; r < 8; ++r) fs[r] = rowScale[wave][r + 8 * half];
    #pragma unroll
    for (int t = 0; t < 16; ++t) {
      #pragma unroll
      for (int r = 0; r < 8; ++r) acc[t][r] *= fs[r];
    }

    // A fragments of P (16x64 bf16 -> two 16x32 fragments)
    FragB aP0, aP1;
    #pragma unroll
    for (int v = 0; v < 8; ++v) {
      int kb = ((v < 4) ? 2 * v : 2 * v + 8) + 8 * half;
      aP0.u[v] = *(const unsigned*)(&Ps[wave][l16 * 64 + kb]);
      aP1.u[v] = *(const unsigned*)(&Ps[wave][l16 * 64 + 32 + kb]);
    }

    // O += P (16x64) x V^T (64x256): 32 bf16 WMMAs over channel tiles
    #pragma unroll
    for (int t = 0; t < 16; ++t) {
      FragB bV0, bV1;
      #pragma unroll
      for (int v = 0; v < 8; ++v) {
        int kb = ((v < 4) ? 2 * v : 2 * v + 8) + 8 * half;
        bV0.u[v] = *(const unsigned*)(Vs + (t * 16 + l16) * 64 + kb);
        bV1.u[v] = *(const unsigned*)(Vs + (t * 16 + l16) * 64 + 32 + kb);
      }
      acc[t] = __builtin_amdgcn_wmma_f32_16x16x32_bf16(false, aP0.v, false, bV0.v, (short)0, acc[t], false, false);
      acc[t] = __builtin_amdgcn_wmma_f32_16x16x32_bf16(false, aP1.v, false, bV1.v, (short)0, acc[t], false, false);
    }
  }

  // epilogue: normalize, residual, store self_out, accumulate channel means
  if (lane < 16) rowLs[wave][lane] = l_i;
  __syncthreads();
  float linv[8];
  #pragma unroll
  for (int r = 0; r < 8; ++r) linv[r] = 1.0f / rowLs[wave][r + 8 * half];

  const float g = gamma[0];
  #pragma unroll
  for (int t = 0; t < 16; ++t) {
    int cg = t * 16 + l16;
    float partial = 0.f;
    #pragma unroll
    for (int r = 0; r < 8; ++r) {
      int mg = mBlock + wave * 16 + r + 8 * half;
      size_t idx = ((size_t)(b * C_DIM + cg)) * NPIX + mg;
      float so = g * (acc[t][r] * linv[r]) + x[idx];
      self_out[idx] = so;
      partial += so;
    }
    atomicAdd(mean_acc + b * C_DIM + cg, partial);
  }
}

// ---------------- SE branch ----------------
__global__ void zero_kernel(float* p, int n) {
  int i = blockIdx.x * 256 + threadIdx.x;
  if (i < n) p[i] = 0.f;
}

__global__ __launch_bounds__(256)
void se_kernel(const float* __restrict__ mean_acc,
               const float* __restrict__ W1, const float* __restrict__ W2,
               float* __restrict__ y)
{
  __shared__ float sm[C_DIM];
  __shared__ float sh[16];
  int b = blockIdx.x, t = threadIdx.x;
  sm[t] = mean_acc[b * C_DIM + t] * (1.0f / (float)NPIX);
  __syncthreads();
  if (t < 16) {
    float h = 0.f;
    for (int j = 0; j < C_DIM; ++j) h += W1[t * C_DIM + j] * sm[j];
    sh[t] = fmaxf(h, 0.f);
  }
  __syncthreads();
  float z = 0.f;
  #pragma unroll
  for (int k = 0; k < 16; ++k) z += W2[t * 16 + k] * sh[k];
  y[b * C_DIM + t] = 1.0f / (1.0f + __expf(-z));
}

__global__ __launch_bounds__(256)
void final_kernel(const float* __restrict__ self_out, const float* __restrict__ y,
                  const float* __restrict__ alpha, const float* __restrict__ beta,
                  float* __restrict__ out)
{
  int i = blockIdx.x * 256 + threadIdx.x;
  int c = (i >> 12) & (C_DIM - 1);
  int b = i >> 20;
  float so = self_out[i];
  float yv = y[b * C_DIM + c];
  out[i] = alpha[0] * so + beta[0] * so * yv;
}

// ---------------- launch ----------------
extern "C" void kernel_launch(void* const* d_in, const int* in_sizes, int n_in,
                              void* d_out, int out_size, void* d_ws, size_t ws_size,
                              hipStream_t stream) {
  const float* x     = (const float*)d_in[0];
  const float* Wq    = (const float*)d_in[1];
  const float* bq    = (const float*)d_in[2];
  const float* Wk    = (const float*)d_in[3];
  const float* bk    = (const float*)d_in[4];
  const float* Wv    = (const float*)d_in[5];
  const float* bv    = (const float*)d_in[6];
  const float* gamma = (const float*)d_in[7];
  const float* W1    = (const float*)d_in[8];
  const float* W2    = (const float*)d_in[9];
  const float* alpha = (const float*)d_in[10];
  const float* beta  = (const float*)d_in[11];

  char* ws = (char*)d_ws;
  unsigned short* qw       = (unsigned short*)(ws);                        // 1 MB
  unsigned short* kw       = (unsigned short*)(ws + (size_t)(1  << 20));   // 1 MB
  unsigned short* vw       = (unsigned short*)(ws + (size_t)(2  << 20));   // 8 MB
  float*          self_out = (float*)         (ws + (size_t)(10 << 20));   // 16 MB
  float*          mean_acc = (float*)         (ws + (size_t)(26 << 20));   // 4 KB
  float*          yv       = (float*)         (ws + (size_t)(26 << 20) + 8192);

  zero_kernel<<<(BATCH * C_DIM + 255) / 256, 256, 0, stream>>>(mean_acc, BATCH * C_DIM);
  proj_qk_kernel<<<(BATCH * NPIX * CQ) / 256, 256, 0, stream>>>(x, Wq, bq, Wk, bk, qw, kw);
  proj_v_kernel<<<(BATCH * C_DIM * NPIX) / 256, 256, 0, stream>>>(x, Wv, bv, vw);
  dim3 g(NPIX / 128, BATCH);
  attn_kernel<<<g, 256, 0, stream>>>(qw, kw, vw, x, gamma, self_out, mean_acc);
  se_kernel<<<BATCH, 256, 0, stream>>>(mean_acc, W1, W2, yv);
  final_kernel<<<(BATCH * C_DIM * NPIX) / 256, 256, 0, stream>>>(self_out, yv, alpha, beta, (float*)d_out);
}
